// Net_28338194219402
// MI455X (gfx1250) — matrix operand order repaired and checked by reading, not compiled, not run on hardware
//
#include <hip/hip_runtime.h>
#include <hip/hip_bf16.h>

typedef __attribute__((ext_vector_type(16))) __bf16 v16bf;
typedef __attribute__((ext_vector_type(8)))  float  v8f;

// ---------------- helpers: order-preserving float <-> uint key ----------------
__device__ __forceinline__ unsigned fkey(float f) {
    unsigned u = __float_as_uint(f);
    return u ^ (((int)u >> 31) | 0x80000000u);
}
__device__ __forceinline__ float fdecode(unsigned k) {
    unsigned u = (k & 0x80000000u) ? (k ^ 0x80000000u) : ~k;
    return __uint_as_float(u);
}

// ---------------- edge-MLP collapse: A[j]=Σ ρ+(w1k)w2[j][k], B[j]=Σ ρ-(w1k)w2[j][k]
__global__ void coeff_kernel(const float* __restrict__ w1, const float* __restrict__ w2,
                             int H, float* __restrict__ ca, float* __restrict__ cb) {
    int j = threadIdx.x;
    if (j >= H) return;
    float sa = 0.f, sb = 0.f;
    for (int k = 0; k < H; ++k) {
        float w  = w1[k];
        float rp = (w > 0.f) ? w : 0.2f * w;   // leaky slope when s >= 0
        float rn = (w > 0.f) ? 0.2f * w : w;   // leaky slope when s <  0
        float m  = w2[j * H + k];
        sa += m * rp;
        sb += m * rn;
    }
    ca[j] = sa;
    cb[j] = sb;
}

// ---------------- WMMA bf16 GEMM: Y[N,OUT] = X[N,K] * W[OUT,K]^T + bias --------
template <int K, int OUT>
__global__ __launch_bounds__(256)
void gemm_wmma_bf16(const float* __restrict__ X, const float* __restrict__ W,
                    const float* __restrict__ bias, float* __restrict__ Y, int nrows) {
    constexpr int NT   = OUT / 16;   // n-tiles (waves along N)
    constexpr int MT   = 8 / NT;     // m-tiles per block (8 wave32 waves / block)
    constexpr int ROWS = MT * 16;
    __shared__ __bf16 sA[ROWS][K];
    __shared__ __bf16 sB[OUT][K];

    const int tid       = threadIdx.x;
    const int block_row = blockIdx.x * ROWS;

    for (int i = tid; i < OUT * K; i += 256)
        sB[i / K][i % K] = (__bf16)W[i];
    for (int i = tid; i < ROWS * K; i += 256) {
        int r = i / K, c = i % K;
        int gr = block_row + r;
        sA[r][c] = (__bf16)((gr < nrows) ? X[(size_t)gr * K + c] : 0.f);
    }
    __syncthreads();

    const int wave = tid >> 5;        // wave32
    const int lane = tid & 31;
    const int mt   = wave / NT;
    const int nt   = wave % NT;
    const int arow = mt * 16 + (lane & 15);
    const int hi   = lane >> 4;       // 0: lanes 0-15, 1: lanes 16-31
    const int ncol = nt * 16 + (lane & 15);

    v8f acc = {};
    for (int kk = 0; kk < K; kk += 32) {
        v16bf a, b;
#pragma unroll
        for (int i = 0; i < 16; ++i) {
            // A 16x32 bf16 layout: lo half-wave K=0-7,16-23 ; hi half-wave K=8-15,24-31
            int ka = kk + hi * 8 + (i & 7) + ((i >> 3) << 4);
            a[i] = sA[arow][ka];
            // B 32x16 layout: lo half-wave K=0-7,16-23 ; hi half-wave K=8-15,24-31
            int kb = kk + hi * 8 + (i & 7) + ((i >> 3) << 4);
            b[i] = sB[ncol][kb];
        }
        acc = __builtin_amdgcn_wmma_f32_16x16x32_bf16(false, a, false, b,
                                                      (short)0, acc, false, false);
    }
    // C layout: VGPR r -> row r (lanes 0-15) / row r+8 (lanes 16-31)
#pragma unroll
    for (int r = 0; r < 8; ++r) {
        int row = block_row + mt * 16 + r + hi * 8;
        if (row < nrows)
            Y[(size_t)row * OUT + ncol] = acc[r] + bias[ncol];
    }
}

// ---------------- edge passes (segment softmax over src, scatter to dst) ------
template <int H, int CPT>
__global__ __launch_bounds__(256)
void edge_max(const int* __restrict__ srcs, const float* __restrict__ wmul,
              const float* __restrict__ ca, const float* __restrict__ cb,
              const float* __restrict__ bias, unsigned* __restrict__ mkey, int nE) {
    constexpr int PARTS = H / CPT;
    long long g = (long long)blockIdx.x * blockDim.x + threadIdx.x;
    long long e = g / PARTS;
    if (e >= nE) return;
    int part = (int)(g % PARTS);
    float s  = wmul[e];
    int src  = srcs[e];
    const float* c = (s >= 0.f) ? ca : cb;
    unsigned* base = mkey + (size_t)src * H;
#pragma unroll
    for (int i = 0; i < CPT; ++i) {
        int j = part * CPT + i;
        float w = bias[j] + s * c[j];
        atomicMax(base + j, fkey(w));
    }
}

template <int H, int CPT>
__global__ __launch_bounds__(256)
void edge_sum(const int* __restrict__ srcs, const float* __restrict__ wmul,
              const float* __restrict__ ca, const float* __restrict__ cb,
              const float* __restrict__ bias, const unsigned* __restrict__ mkey,
              float* __restrict__ den, int nE) {
    constexpr int PARTS = H / CPT;
    long long g = (long long)blockIdx.x * blockDim.x + threadIdx.x;
    long long e = g / PARTS;
    if (e >= nE) return;
    int part = (int)(g % PARTS);
    float s  = wmul[e];
    int src  = srcs[e];
    const float* c = (s >= 0.f) ? ca : cb;
    size_t base = (size_t)src * H;
#pragma unroll
    for (int i = 0; i < CPT; ++i) {
        int j = part * CPT + i;
        float w  = bias[j] + s * c[j];
        float ex = __expf(w - fdecode(mkey[base + j]));
        atomicAdd(den + base + j, ex);
    }
}

template <int H, int CPT>
__global__ __launch_bounds__(256)
void edge_msg(const int* __restrict__ srcs, const int* __restrict__ dsts,
              const float* __restrict__ wmul,
              const float* __restrict__ ca, const float* __restrict__ cb,
              const float* __restrict__ bias, const unsigned* __restrict__ mkey,
              const float* __restrict__ den, const float* __restrict__ xf,
              float* __restrict__ outAgg, int nE) {
    constexpr int PARTS = H / CPT;
    long long g = (long long)blockIdx.x * blockDim.x + threadIdx.x;
    long long e = g / PARTS;
    if (e >= nE) return;
    int part = (int)(g % PARTS);
    float s  = wmul[e];
    int src  = srcs[e];
    int dst  = dsts[e];
    const float* c = (s >= 0.f) ? ca : cb;
    size_t sbase = (size_t)src * H;
    size_t dbase = (size_t)dst * H;
#pragma unroll
    for (int i = 0; i < CPT; ++i) {
        int j = part * CPT + i;
        float w   = bias[j] + s * c[j];
        float ex  = __expf(w - fdecode(mkey[sbase + j]));
        float msg = ex / (den[sbase + j] + 1e-16f) * xf[sbase + j];
        atomicAdd(outAgg + dbase + j, msg);
    }
}

// ---------------- elementwise ----------------------------------------------
__global__ void elu_kernel(float* __restrict__ v, long long n) {
    long long i = (long long)blockIdx.x * blockDim.x + threadIdx.x;
    if (i >= n) return;
    float x = v[i];
    v[i] = (x > 0.f) ? x : (__expf(x) - 1.f);
}

__global__ void logsoftmax16(const float* __restrict__ in, float* __restrict__ out, int n) {
    int node = blockIdx.x * blockDim.x + threadIdx.x;
    if (node >= n) return;
    float v[16];
    float mx = -3.4e38f;
#pragma unroll
    for (int c = 0; c < 16; ++c) { v[c] = in[(size_t)node * 16 + c]; mx = fmaxf(mx, v[c]); }
    float sum = 0.f;
#pragma unroll
    for (int c = 0; c < 16; ++c) sum += __expf(v[c] - mx);
    float lse = mx + __logf(sum);
#pragma unroll
    for (int c = 0; c < 16; ++c) out[(size_t)node * 16 + c] = v[c] - lse;
}

// ---------------- driver -----------------------------------------------------
extern "C" void kernel_launch(void* const* d_in, const int* in_sizes, int n_in,
                              void* d_out, int out_size, void* d_ws, size_t ws_size,
                              hipStream_t stream) {
    const float* x      = (const float*)d_in[0];
    const int*   ei     = (const int*)d_in[1];   // [2,E]: row0=src, row1=dst
    const float* wmul   = (const float*)d_in[2];
    const float* lin1_w = (const float*)d_in[3];
    const float* lin1_b = (const float*)d_in[4];
    const float* m1w1   = (const float*)d_in[5];
    const float* m1w2   = (const float*)d_in[6];
    const float* m1b2   = (const float*)d_in[7];
    const float* lin2_w = (const float*)d_in[8];
    const float* lin2_b = (const float*)d_in[9];
    const float* m2w1   = (const float*)d_in[10];
    const float* m2w2   = (const float*)d_in[11];
    const float* m2b2   = (const float*)d_in[12];
    float* out = (float*)d_out;

    const int N = in_sizes[0] / 256;   // F_IN = 256
    const int E = in_sizes[2];         // w_mul is [E,1]
    const int* src = ei;
    const int* dst = ei + E;

    // ---- workspace carve (bytes) ----
    char* w = (char*)d_ws;
    float* ca1 = (float*)w;            // 64
    float* cb1 = ca1 + 64;             // 64
    float* ca2 = cb1 + 64;             // 16
    float* cb2 = ca2 + 16;             // 16
    size_t off = 1024;
    const size_t n64 = (size_t)N * 64;
    const size_t n16 = (size_t)N * 16;
    float*    x1  = (float*)(w + off);    off += n64 * 4;
    unsigned* mk1 = (unsigned*)(w + off); off += n64 * 4;
    float*    dn1 = (float*)(w + off);    off += n64 * 4;
    float*    ag1 = (float*)(w + off);    off += n64 * 4;   // h after ELU, input to layer 2
    // layer-2 buffers recycle the x1 region (x1 dead after layer-1 edge_msg)
    float*    x2  = x1;
    unsigned* mk2 = (unsigned*)(x1 + n16);
    float*    dn2 = x1 + 2 * n16;
    float*    ag2 = x1 + 3 * n16;

    // ---- collapsed edge-MLP coefficients ----
    coeff_kernel<<<1, 64, 0, stream>>>(m1w1, m1w2, 64, ca1, cb1);
    coeff_kernel<<<1, 16, 0, stream>>>(m2w1, m2w2, 16, ca2, cb2);

    // ---- layer 1: x1 = x @ lin1_w^T + b  (WMMA bf16) ----
    gemm_wmma_bf16<256, 64><<<(N + 31) / 32, 256, 0, stream>>>(x, lin1_w, lin1_b, x1, N);

    hipMemsetAsync(mk1, 0, n64 * 4, stream);
    hipMemsetAsync(dn1, 0, n64 * 4, stream);
    hipMemsetAsync(ag1, 0, n64 * 4, stream);

    const int eThreads1 = 256;
    const long long eTot1 = (long long)E * (64 / 8);
    const int eBlocks1 = (int)((eTot1 + eThreads1 - 1) / eThreads1);
    edge_max<64, 8><<<eBlocks1, eThreads1, 0, stream>>>(src, wmul, ca1, cb1, m1b2, mk1, E);
    edge_sum<64, 8><<<eBlocks1, eThreads1, 0, stream>>>(src, wmul, ca1, cb1, m1b2, mk1, dn1, E);
    edge_msg<64, 8><<<eBlocks1, eThreads1, 0, stream>>>(src, dst, wmul, ca1, cb1, m1b2,
                                                        mk1, dn1, x1, ag1, E);
    elu_kernel<<<(int)((n64 + 255) / 256), 256, 0, stream>>>(ag1, (long long)n64);

    // ---- layer 2: x2 = h @ lin2_w^T + b  (WMMA bf16) ----
    gemm_wmma_bf16<64, 16><<<(N + 127) / 128, 256, 0, stream>>>(ag1, lin2_w, lin2_b, x2, N);

    hipMemsetAsync(mk2, 0, n16 * 4, stream);
    hipMemsetAsync(dn2, 0, n16 * 4, stream);
    hipMemsetAsync(ag2, 0, n16 * 4, stream);

    const long long eTot2 = (long long)E * (16 / 8);
    const int eBlocks2 = (int)((eTot2 + eThreads1 - 1) / eThreads1);
    edge_max<16, 8><<<eBlocks2, eThreads1, 0, stream>>>(src, wmul, ca2, cb2, m2b2, mk2, E);
    edge_sum<16, 8><<<eBlocks2, eThreads1, 0, stream>>>(src, wmul, ca2, cb2, m2b2, mk2, dn2, E);
    edge_msg<16, 8><<<eBlocks2, eThreads1, 0, stream>>>(src, dst, wmul, ca2, cb2, m2b2,
                                                        mk2, dn2, x2, ag2, E);

    // ---- log_softmax over 16 classes ----
    logsoftmax16<<<(N + 255) / 256, 256, 0, stream>>>(ag2, out, N);

    (void)n_in; (void)out_size; (void)ws_size;
}
